// Hierarch_RNN_66228395704903
// MI455X (gfx1250) — matrix operand order repaired and checked by reading, not compiled, not run on hardware
//
#include <hip/hip_runtime.h>
#include <hip/hip_bf16.h>
#include <math.h>

typedef __bf16 bf16;
typedef __attribute__((ext_vector_type(16))) __bf16 v16bf;
typedef __attribute__((ext_vector_type(8)))  __bf16 v8bf;
typedef __attribute__((ext_vector_type(8)))  float  v8f;

#define NBC   10272          // B*C = 32*321
#define DDIM  256
#define G3    768
#define SXN   30
#define SEGN  24
#define LIN   720
#define CCH   321
#define MWG   16             // rows (sequences) per workgroup in k_gru

// -------- workspace layout (bytes, all 256B aligned) --------
#define OFF_XEMB   ((size_t)0)
#define SZ_XEMB    ((size_t)SXN*NBC*DDIM*2)          // 157,777,920
#define OFF_WIH    (OFF_XEMB + SZ_XEMB)
#define SZ_W       ((size_t)G3*DDIM*2)               // 393,216
#define OFF_WHH    (OFF_WIH + SZ_W)
#define OFF_POSG   (OFF_WHH + SZ_W)
#define SZ_POSG    ((size_t)SXN*G3*4)
#define OFF_CHANG  (OFF_POSG + SZ_POSG)
#define SZ_CHANG   ((size_t)CCH*G3*4)
#define OFF_HT     (OFF_CHANG + SZ_CHANG)
#define SZ_HT      ((size_t)NBC*DDIM*4)
#define OFF_GH     (OFF_HT + SZ_HT)

// ---------------- helpers ----------------
__device__ __forceinline__ float sigmoidf_(float v){ return 1.f/(1.f+__expf(-v)); }

__device__ __forceinline__ v16bf cat8(v8bf lo, v8bf hi){
  return __builtin_shufflevector(lo,hi,0,1,2,3,4,5,6,7,8,9,10,11,12,13,14,15);
}
// A-operand (16x32 bf16): per lane, 16 contiguous halves
__device__ __forceinline__ v16bf ldA(const bf16* p){
  v8bf lo = *(const v8bf*)p;
  v8bf hi = *(const v8bf*)(p + 8);
  return cat8(lo, hi);
}
#define WMMA_BF16(A,B,C) __builtin_amdgcn_wmma_f32_16x16x32_bf16(false,(A),false,(B),(short)0,(C),false,false)

// ---------------- kernel 1: fp32 -> bf16 weight conversion ----------------
__global__ void k_wcvt(const float* __restrict__ wih, const float* __restrict__ whh,
                       bf16* __restrict__ wihb, bf16* __restrict__ whhb) {
  int i = blockIdx.x*blockDim.x + threadIdx.x;
  if (i < G3*DDIM) { wihb[i] = (bf16)wih[i]; whhb[i] = (bf16)whh[i]; }
}

// ---------------- kernel 2: pos_g / chan_g (pe @ w_ih^T split) ----------------
__global__ void k_posg(const float* __restrict__ pos, const float* __restrict__ chan,
                       const float* __restrict__ wih, const float* __restrict__ bih,
                       float* __restrict__ pos_g, float* __restrict__ chan_g) {
  int i = blockIdx.x*blockDim.x + threadIdx.x;
  if (i < SXN*G3) {
    int s = i / G3, g = i % G3;
    float acc = 0.f;
    for (int k = 0; k < 128; ++k) acc += pos[s*128+k]*wih[g*DDIM+k];
    pos_g[i] = acc;
  } else {
    i -= SXN*G3;
    if (i < CCH*G3) {
      int c = i / G3, g = i % G3;
      float acc = bih[g];                         // fold b_ih here
      for (int k = 0; k < 128; ++k) acc += chan[c*128+k]*wih[g*DDIM+128+k];
      chan_g[i] = acc;
    }
  }
}

// ---------------- kernel 3: xemb = relu(xseg @ emb_w^T + b) -> bf16 ----------------
__global__ void k_emb(const float* __restrict__ x, const float* __restrict__ ew,
                      const float* __restrict__ eb, bf16* __restrict__ xemb) {
  const int bid = blockIdx.x;             // t*NBC + bc
  const int t = bid / NBC, bc = bid % NBC;
  const int b = bc / CCH, c = bc % CCH;
  __shared__ float xv[SEGN];
  const int tid = threadIdx.x;
  if (tid < SEGN) {
    const int l = t*SEGN + tid;
    xv[tid] = x[((size_t)b*LIN + l)*CCH + c] - x[((size_t)b*LIN + (LIN-1))*CCH + c];
  }
  __syncthreads();
  float s = eb[tid];
  #pragma unroll
  for (int j = 0; j < SEGN; ++j) s += ew[tid*SEGN + j]*xv[j];
  s = fmaxf(s, 0.f);
  xemb[((size_t)t*NBC + bc)*DDIM + tid] = (bf16)s;
}

// ---------------- kernel 4: persistent fused GRU scan (WMMA bf16) ----------------
// Block = 512 threads = 16 waves; each block owns 16 rows (sequences).
// Wave nt computes the 16-row x 16-col gate tiles at columns n0, n0+256, n0+512
// for both gi (A = xemb tile) and gh (A = h tile): 6 f32 accumulators per wave.
// Weights stream from global/L2 every step. An opaque *integer* offset (zero at
// runtime) is injected per iteration: it blocks LLVM from hoisting/CSE'ing the
// t-invariant weight loads (which previously caused scratch spills) while
// preserving addrspace(1) inference so the loads stay global_load_b128
// (a raw opaque pointer degraded them to flat_load, polluting DScnt).
__global__ __launch_bounds__(512, 1)
void k_gru(const bf16* __restrict__ xemb, const bf16* __restrict__ wih, const bf16* __restrict__ whh,
           const float* __restrict__ bih, const float* __restrict__ bhh,
           float* __restrict__ hT, float* __restrict__ gh) {
  __shared__ __align__(16) float h_f[MWG*DDIM];   // fp32 master state
  __shared__ __align__(32) bf16  h_b[MWG*DDIM];   // bf16 copy for WMMA A
  const int tid   = threadIdx.x;
  const int nt    = tid >> 5;                     // wave id 0..15 -> column tile
  const int lane  = tid & 31;
  const int half  = lane >> 4, colL = lane & 15;
  const int col   = nt*16 + colL;
  const int rowlA = colL;                         // A-operand row (M = lane%16)
  const int rowbase = blockIdx.x * MWG;

  for (int i = tid; i < MWG*DDIM; i += 512) { h_f[i] = 0.f; h_b[i] = (bf16)0.f; }

  const float bihr = bih[col], bihz = bih[256+col], bihn = bih[512+col];
  const float bhhr = bhh[col], bhhz = bhh[256+col], bhhn = bhh[512+col];
  const bf16* wi0 = wih + (size_t)col*DDIM;       // B[k][n] = W[n][k] (row-major W)
  const bf16* wh0 = whh + (size_t)col*DDIM;

  for (int t = 0; t < SXN; ++t) {
    __syncthreads();                              // h writes from prev step visible
    int toff = 0;                                 // opaque 0: defeats hoisting only
    asm volatile("" : "+v"(toff));
    const bf16* wi_t = wi0 + toff;
    const bf16* wh_t = wh0 + toff;
    v8f ir={},iz={},inn={},hr={},hz={},hn={};
    const bf16* xr = xemb + ((size_t)t*NBC + rowbase + rowlA)*DDIM;
    #pragma unroll 2
    for (int kc = 0; kc < 8; ++kc) {
      const int ka = kc*32 + 16*half;             // A chunk offset
      const int kb = kc*32 + 16*half;             // B chunk offset
      v16bf ax = ldA(xr + ka);
      v16bf ah = ldA(&h_b[rowlA*DDIM + ka]);
      v16bf b0 = *(const v16bf*)(wi_t + kb);
      v16bf b1 = *(const v16bf*)(wi_t + (size_t)256*DDIM + kb);
      v16bf b2 = *(const v16bf*)(wi_t + (size_t)512*DDIM + kb);
      v16bf c0 = *(const v16bf*)(wh_t + kb);
      v16bf c1 = *(const v16bf*)(wh_t + (size_t)256*DDIM + kb);
      v16bf c2 = *(const v16bf*)(wh_t + (size_t)512*DDIM + kb);
      ir  = WMMA_BF16(ax, b0, ir);
      iz  = WMMA_BF16(ax, b1, iz);
      inn = WMMA_BF16(ax, b2, inn);
      hr  = WMMA_BF16(ah, c0, hr);
      hz  = WMMA_BF16(ah, c1, hz);
      hn  = WMMA_BF16(ah, c2, hn);
    }
    if (t + 1 < SXN) {                            // prefetch next step's A rows
      __builtin_prefetch(xemb + ((size_t)(t+1)*NBC + rowbase + rowlA)*DDIM, 0, 3);
    }
    __syncthreads();                              // all h_b reads done before rewrite
    #pragma unroll
    for (int v = 0; v < 8; ++v) {                 // D row = v + 8*half, col = lane%16
      const int rowl = v + 8*half;
      const float xt   = (float)xemb[((size_t)t*NBC + rowbase + rowl)*DDIM + col];
      const float hold = h_f[rowl*DDIM + col];
      const float r  = sigmoidf_(ir[v]+bihr + hr[v]+bhhr);
      const float z  = sigmoidf_(iz[v]+bihz + hz[v]+bhhz);
      const float nn = tanhf(inn[v]+bihn + r*(hn[v]+bhhn));
      const float hnew = xt + (1.f - z)*nn + z*hold;   // residual GRU
      h_f[rowl*DDIM + col] = hnew;
      h_b[rowl*DDIM + col] = (bf16)hnew;
    }
  }

  // tail: gh = hT @ w_hh^T + b_hh (needed once for the decode step), dump hT
  __syncthreads();
  v8f thr={},thz={},thn={};
  #pragma unroll 2
  for (int kc = 0; kc < 8; ++kc) {
    const int ka = kc*32 + 16*half;
    const int kb = kc*32 + 16*half;
    v16bf ah = ldA(&h_b[rowlA*DDIM + ka]);
    thr = WMMA_BF16(ah, *(const v16bf*)(wh0 + kb), thr);
    thz = WMMA_BF16(ah, *(const v16bf*)(wh0 + (size_t)256*DDIM + kb), thz);
    thn = WMMA_BF16(ah, *(const v16bf*)(wh0 + (size_t)512*DDIM + kb), thn);
  }
  #pragma unroll
  for (int v = 0; v < 8; ++v) {
    const int rowl = v + 8*half;
    const size_t rowg = (size_t)rowbase + rowl;
    gh[rowg*G3 + col]       = thr[v] + bhhr;
    gh[rowg*G3 + 256 + col] = thz[v] + bhhz;
    gh[rowg*G3 + 512 + col] = thn[v] + bhhn;
    hT[rowg*DDIM + col]     = h_f[rowl*DDIM + col];
  }
}

// ---------------- kernel 5: decode GRU + pred + scrambled scatter ----------------
__global__ void k_fin(const float* __restrict__ pos_g, const float* __restrict__ chan_g,
                      const float* __restrict__ gh, const float* __restrict__ hT,
                      const float* __restrict__ pw, const float* __restrict__ pb,
                      const float* __restrict__ x, float* __restrict__ out) {
  const int r  = blockIdx.x;                 // r = s2*NBC + bc  (pe flat row)
  const int bc = r % NBC;
  const int rem = r % (CCH*SXN);             // tile-then-reshape scramble
  const int c_p = rem / SXN, s_p = rem % SXN;
  __shared__ float hy[DDIM];
  __shared__ float part[8*24];
  const int d = threadIdx.x;
  {
    const float gir = pos_g[s_p*G3 + d]       + chan_g[c_p*G3 + d];
    const float giz = pos_g[s_p*G3 + 256 + d] + chan_g[c_p*G3 + 256 + d];
    const float gin = pos_g[s_p*G3 + 512 + d] + chan_g[c_p*G3 + 512 + d];
    const float ghr = gh[(size_t)bc*G3 + d];
    const float ghz = gh[(size_t)bc*G3 + 256 + d];
    const float ghn = gh[(size_t)bc*G3 + 512 + d];
    const float h   = hT[(size_t)bc*DDIM + d];
    const float rr = sigmoidf_(gir + ghr);
    const float zz = sigmoidf_(giz + ghz);
    const float nn = tanhf(gin + rr*ghn);
    hy[d] = (1.f - zz)*nn + zz*h;
  }
  __syncthreads();
  if (d < 192) {
    const int p = d / 24, k = d % 24;
    float acc = 0.f;
    #pragma unroll
    for (int j = 0; j < 32; ++j) acc += pw[k*DDIM + p*32 + j]*hy[p*32 + j];
    part[p*24 + k] = acc;
  }
  __syncthreads();
  if (d < 24) {
    float y = pb[d];
    #pragma unroll
    for (int p = 0; p < 8; ++p) y += part[p*24 + d];
    const size_t f = (size_t)r*24 + d;       // flat index of y before reshape
    const int b_o  = (int)(f / ((size_t)CCH*LIN));
    const int rem2 = (int)(f % ((size_t)CCH*LIN));
    const int c_o = rem2 / LIN, l_o = rem2 % LIN;
    out[((size_t)b_o*LIN + l_o)*CCH + c_o] =
        y + x[((size_t)b_o*LIN + (LIN-1))*CCH + c_o];   // + seq_last
  }
}

extern "C" void kernel_launch(void* const* d_in, const int* in_sizes, int n_in,
                              void* d_out, int out_size, void* d_ws, size_t ws_size,
                              hipStream_t stream) {
  (void)in_sizes; (void)n_in; (void)out_size; (void)ws_size;
  // inputs: 0=x; layer0 params 1..10 are dead code; layer1: 11..20
  const float* x    = (const float*)d_in[0];
  const float* ew   = (const float*)d_in[11];   // emb_w1 (256,24)
  const float* eb   = (const float*)d_in[12];   // emb_b1 (256)
  const float* wih  = (const float*)d_in[13];   // gru_wih1 (768,256)
  const float* whh  = (const float*)d_in[14];   // gru_whh1 (768,256)
  const float* bih  = (const float*)d_in[15];   // (768)
  const float* bhh  = (const float*)d_in[16];   // (768)
  const float* pw   = (const float*)d_in[17];   // pred_w1 (24,256)
  const float* pb   = (const float*)d_in[18];   // pred_b1 (24)
  const float* pos  = (const float*)d_in[19];   // pos1 (30,128)
  const float* chan = (const float*)d_in[20];   // chan1 (321,128)

  char* ws = (char*)d_ws;
  bf16*  xemb  = (bf16*)(ws + OFF_XEMB);
  bf16*  wihb  = (bf16*)(ws + OFF_WIH);
  bf16*  whhb  = (bf16*)(ws + OFF_WHH);
  float* posg  = (float*)(ws + OFF_POSG);
  float* chang = (float*)(ws + OFF_CHANG);
  float* hT    = (float*)(ws + OFF_HT);
  float* gh    = (float*)(ws + OFF_GH);
  float* out   = (float*)d_out;

  hipLaunchKernelGGL(k_wcvt, dim3((G3*DDIM+255)/256), dim3(256), 0, stream,
                     wih, whh, wihb, whhb);
  hipLaunchKernelGGL(k_posg, dim3(((SXN+CCH)*G3+255)/256), dim3(256), 0, stream,
                     pos, chan, wih, bih, posg, chang);
  hipLaunchKernelGGL(k_emb, dim3(SXN*NBC), dim3(256), 0, stream, x, ew, eb, xemb);
  hipLaunchKernelGGL(k_gru, dim3(NBC/MWG), dim3(512), 0, stream,
                     xemb, wihb, whhb, bih, bhh, hT, gh);
  hipLaunchKernelGGL(k_fin, dim3(SXN*NBC), dim3(256), 0, stream,
                     posg, chang, gh, hT, pw, pb, x, out);
}
